// MultiHeadAttention_84104049590754
// MI455X (gfx1250) — matrix operand order repaired and checked by reading, not compiled, not run on hardware
//
#include <hip/hip_runtime.h>
#include <hip/hip_bf16.h>

// MHA forward for MI455X (gfx1250): bf16 WMMA (16x16x32), fp32 accum,
// flash-style online softmax, TDM (tensor_load_to_lds) for K/V tile staging,
// LDS-staged coalesced b128 stores. wave32 layouts per CDNA5 ISA.

#define NHEADS 16
#define HS     64
#define BATCH  4
#define SEQ    2048
#define VDIM   1024
#define TOKENS (BATCH * SEQ)   // 8192

typedef __attribute__((ext_vector_type(16))) __bf16          v16bf;
typedef __attribute__((ext_vector_type(16))) unsigned short  v16us;
typedef __attribute__((ext_vector_type(8)))  float           v8f;
typedef __attribute__((ext_vector_type(4)))  unsigned int    u32x4;
typedef __attribute__((ext_vector_type(8)))  unsigned int    u32x8;

union Frag16 {
    v16bf bf;
    v16us us;
};

__device__ __forceinline__ unsigned short f2bf(float f) {
    unsigned int u = __float_as_uint(f);
    unsigned int r = u + 0x7FFFu + ((u >> 16) & 1u);   // round-to-nearest-even
    return (unsigned short)(r >> 16);
}

__device__ __forceinline__ v8f wmma_bf16(const Frag16& a, const Frag16& b, v8f c) {
    // (neg_a, A, neg_b, B, c_mod, C, reuse_a, reuse_b)
    return __builtin_amdgcn_wmma_f32_16x16x32_bf16(false, a.bf, false, b.bf,
                                                   (short)0, c, false, false);
}

__device__ __forceinline__ float rmax16(float v) {
#pragma unroll
    for (int m = 1; m < 16; m <<= 1) v = fmaxf(v, __shfl_xor(v, m));
    return v;
}
__device__ __forceinline__ float rsum16(float v) {
#pragma unroll
    for (int m = 1; m < 16; m <<= 1) v += __shfl_xor(v, m);
    return v;
}

// ---------------------------------------------------------------------------
// TDM: 1-D contiguous tile (ndwords DWORDs) from global -> LDS.
// D# per CDNA5 ISA: group0 (4 SGPRs) + group1 (8 SGPRs).
// Issued wave-wide (EXEC ignored); completion via TENSORcnt.
// ---------------------------------------------------------------------------
__device__ __forceinline__ void tdm_load_1d(unsigned lds_off, const void* gptr,
                                            unsigned ndwords) {
    unsigned long long ga = (unsigned long long)(uintptr_t)gptr;
    u32x4 g0;
    g0[0] = 1u;                                   // count=1, user descriptor
    g0[1] = lds_off;                              // lds_addr [63:32]
    g0[2] = (unsigned)ga;                         // global_addr [95:64]
    g0[3] = (unsigned)((ga >> 32) & 0x01FFFFFFull) | (2u << 30); // [120:96], type=2
    u32x8 g1;
    g1[0] = 2u << 16;                             // data_size = 4 bytes
    g1[1] = ndwords << 16;                        // tensor_dim0 (low 16) @ [79:48]
    g1[2] = 0u;                                   // tensor_dim0 hi / tensor_dim1
    g1[3] = ndwords << 16;                        // tile_dim0 @ [127:112]
    g1[4] = 0u;                                   // tile_dim1=0 (1-D), tile_dim2=0
    g1[5] = 0u; g1[6] = 0u; g1[7] = 0u;           // strides unused for 1-D
    asm volatile("tensor_load_to_lds %0, %1" :: "s"(g0), "s"(g1) : "memory");
}

// ---------------------------------------------------------------------------
// Kernel 0: fp32 -> bf16 conversion (W_out)
// ---------------------------------------------------------------------------
__global__ void cvt_bf16_kernel(const float* __restrict__ src,
                                unsigned short* __restrict__ dst, int n) {
    int i = (blockIdx.x * blockDim.x + threadIdx.x) * 4;
    if (i + 3 < n) {
        float4 v = *(const float4*)(src + i);
        dst[i + 0] = f2bf(v.x);
        dst[i + 1] = f2bf(v.y);
        dst[i + 2] = f2bf(v.z);
        dst[i + 3] = f2bf(v.w);
    }
}

// ---------------------------------------------------------------------------
// Kernel 1: per-head QKV projection (bf16 WMMA), LDS-staged coalesced stores.
//   Q/K/V stored bf16, layout [B][H][S][64]. One wave = (head, 16-token tile).
// ---------------------------------------------------------------------------
__global__ __launch_bounds__(256) void qkv_proj_kernel(
        const float* __restrict__ x, const float* __restrict__ Wqkv,
        unsigned short* __restrict__ Qb, unsigned short* __restrict__ Kb,
        unsigned short* __restrict__ Vb) {
    __shared__ __align__(16) unsigned short sW[192 * 64];       // 24 KB
    __shared__ __align__(16) unsigned short sOut[8][16 * 192];  // 48 KB

    int tid = threadIdx.x;
    for (int i = tid; i < 192 * 64; i += 256) sW[i] = f2bf(Wqkv[i]);
    __syncthreads();

    int lane = tid & 31;
    int wave = tid >> 5;
    int task = blockIdx.x * 8 + wave;    // 16 heads * 512 tiles = 8192 tasks
    int h    = task >> 9;
    int mt   = task & 511;
    int m0   = mt * 16;

    int lm  = lane & 15;
    int hi  = lane >> 4;
    int kb  = hi ? 8 : 0;     // unused; A loads also use kb2
    int kb2 = hi ? 16 : 0;    // B-layout K base

    // A fragments: X tile 16 tokens x 64 dims (two K-steps of 32)
    Frag16 a[2];
    const float* xrow = x + (size_t)(m0 + lm) * VDIM + h * HS;
#pragma unroll
    for (int ks = 0; ks < 2; ++ks) {
        int base = ks * 32 + kb2;
#pragma unroll
        for (int i = 0; i < 8; ++i) a[ks].us[i]     = f2bf(xrow[base + i]);
#pragma unroll
        for (int i = 0; i < 8; ++i) a[ks].us[8 + i] = f2bf(xrow[base + 8 + i]);
    }

    unsigned short* sow     = sOut[wave];
    unsigned short* so_lane = sow + hi * 8 * 192 + lm;   // fixed lane base

#pragma unroll
    for (int et = 0; et < 12; ++et) {        // 12 N-tiles of 16 -> 192 outputs
        v8f c;
#pragma unroll
        for (int i = 0; i < 8; ++i) c[i] = 0.0f;

        Frag16 b0, b1;
        const unsigned short* wrow = &sW[(et * 16 + lm) * 64];
#pragma unroll
        for (int i = 0; i < 16; ++i) b0.us[i] = wrow[kb2 + i];          // K 0..31
#pragma unroll
        for (int i = 0; i < 16; ++i) b1.us[i] = wrow[32 + kb2 + i];     // K 32..63
        c = wmma_bf16(a[0], b0, c);
        c = wmma_bf16(a[1], b1, c);

        // stage C fragment in LDS: row (r+8*hi), col et*16+lm (immediate offsets)
#pragma unroll
        for (int r = 0; r < 8; ++r) so_lane[r * 192 + et * 16] = f2bf(c[r]);
    }

    __builtin_amdgcn_wave_barrier();
    asm volatile("" ::: "memory");

    // coalesced copy-out: 16 tokens x {Q,K,V} x 128B; lane owns token lm,
    // variant v = j*2+hi encodes (dst, 16B chunk).
    int tok = m0 + lm;
    int bb  = tok >> 11;                 // / SEQ
    int ss  = tok & 2047;
    size_t rowbase = (((size_t)bb * NHEADS + h) * SEQ + ss) * HS;
#pragma unroll
    for (int j = 0; j < 12; ++j) {
        int v     = j * 2 + hi;          // 0..23
        int which = v >> 3;              // 0=Q 1=K 2=V
        int ch    = v & 7;               // 16B chunk within 128B row
        uint4 d = *(const uint4*)(sow + lm * 192 + v * 8);
        unsigned short* dstp = (which == 0) ? Qb : ((which == 1) ? Kb : Vb);
        *(uint4*)(dstp + rowbase + ch * 8) = d;
    }
}

// ---------------------------------------------------------------------------
// Kernel 2: causal flash attention per (b, h, 64-query block).
// 4 waves / block; K/V 32x64 bf16 tiles DMA'd into LDS by the TDM.
// ---------------------------------------------------------------------------
__global__ __launch_bounds__(128) void attn_kernel(
        const unsigned short* __restrict__ Qb, const unsigned short* __restrict__ Kb,
        const unsigned short* __restrict__ Vb, unsigned short* __restrict__ Ob) {
    __shared__ __align__(16) unsigned short sK[32 * 64];       // 4 KB
    __shared__ __align__(16) unsigned short sV[32 * 64];       // 4 KB
    __shared__ __align__(16) unsigned short sP[4][16 * 32];    // P C->A relayout
    __shared__ __align__(16) unsigned short sO[4][16 * 64];    // store staging

    int tid  = threadIdx.x;
    int lane = tid & 31;
    int wave = tid >> 5;

    int blk  = blockIdx.x;          // B*H*32 = 2048
    int qblk = blk & 31;
    int bh   = blk >> 5;
    int h    = bh & 15;
    int b    = bh >> 4;

    const unsigned short* Qh = Qb + ((size_t)b * NHEADS + h) * SEQ * HS;
    const unsigned short* Kh = Kb + ((size_t)b * NHEADS + h) * SEQ * HS;
    const unsigned short* Vh = Vb + ((size_t)b * NHEADS + h) * SEQ * HS;

    int q0  = qblk * 64 + wave * 16;
    int lm  = lane & 15;
    int hi  = lane >> 4;
    int kb  = hi ? 8 : 0;
    int kb2 = hi ? 16 : 0;

    // Q A-fragments (16 queries x 64 dims)
    Frag16 qf[2];
    const unsigned short* qrow = Qh + (size_t)(q0 + lm) * HS;
#pragma unroll
    for (int ks = 0; ks < 2; ++ks) {
        int base = ks * 32 + kb2;
#pragma unroll
        for (int i = 0; i < 8; ++i) qf[ks].us[i]     = qrow[base + i];
#pragma unroll
        for (int i = 0; i < 8; ++i) qf[ks].us[8 + i] = qrow[base + 8 + i];
    }

    v8f o0, o1, o2, o3;
    float mrun[8], lrun[8];
#pragma unroll
    for (int i = 0; i < 8; ++i) { o0[i] = o1[i] = o2[i] = o3[i] = 0.0f; }
#pragma unroll
    for (int r = 0; r < 8; ++r) { mrun[r] = -1e30f; lrun[r] = 0.0f; }

    unsigned short* sPw = sP[wave];
    const float scale = 0.125f;   // 1/sqrt(64)
    int kend = qblk * 64 + 64;

    for (int kt = 0; kt < kend; kt += 32) {
        __syncthreads();          // everyone done reading previous K/V tiles
        if (wave == 0) {
            // DMA 32x64 bf16 (4 KB contiguous) tiles into LDS via TDM
            tdm_load_1d((unsigned)(size_t)&sK[0], Kh + (size_t)kt * HS, 1024);
            tdm_load_1d((unsigned)(size_t)&sV[0], Vh + (size_t)kt * HS, 1024);
            __builtin_amdgcn_s_wait_tensorcnt(0);
        }
        __syncthreads();          // tiles visible to all waves

        if (kt <= q0 + 15) {      // some keys in this tile visible to this wave
            // ---- scores: 16 queries x 32 keys (two 16-key groups) ----
            v8f s0, s1;
#pragma unroll
            for (int i = 0; i < 8; ++i) { s0[i] = 0.0f; s1[i] = 0.0f; }
#pragma unroll
            for (int g = 0; g < 2; ++g) {
                Frag16 b0, b1;
                const unsigned short* krow = &sK[(g * 16 + lm) * 64];
#pragma unroll
                for (int i = 0; i < 16; ++i) b0.us[i] = krow[kb2 + i];
#pragma unroll
                for (int i = 0; i < 16; ++i) b1.us[i] = krow[32 + kb2 + i];
                if (g == 0) {
                    s0 = wmma_bf16(qf[0], b0, s0);
                    s0 = wmma_bf16(qf[1], b1, s0);
                } else {
                    s1 = wmma_bf16(qf[0], b0, s1);
                    s1 = wmma_bf16(qf[1], b1, s1);
                }
            }

            // ---- online softmax (rows spread over 16-lane halves) ----
#pragma unroll
            for (int r = 0; r < 8; ++r) {
                int q = q0 + r + 8 * hi;
                float v0 = s0[r] * scale; if (kt + lm      > q) v0 = -1e30f;
                float v1 = s1[r] * scale; if (kt + 16 + lm > q) v1 = -1e30f;
                float tmax = rmax16(fmaxf(v0, v1));
                float mnew = fmaxf(mrun[r], tmax);
                float corr = __expf(mrun[r] - mnew);
                float p0 = __expf(v0 - mnew);
                float p1 = __expf(v1 - mnew);
                float tsum = rsum16(p0 + p1);
                lrun[r] = lrun[r] * corr + tsum;
                mrun[r] = mnew;
                o0[r] *= corr; o1[r] *= corr; o2[r] *= corr; o3[r] *= corr;
                int prow = r + 8 * hi;
                sPw[prow * 32 + lm]      = f2bf(p0);
                sPw[prow * 32 + 16 + lm] = f2bf(p1);
            }
            __builtin_amdgcn_wave_barrier();   // LDS in-order per wave

            // P in A-layout: row = lane%16, K = 32 keys
            Frag16 pf;
            const unsigned short* prow = &sPw[lm * 32];
#pragma unroll
            for (int i = 0; i < 8; ++i) pf.us[i]     = prow[kb2 + i];
#pragma unroll
            for (int i = 0; i < 8; ++i) pf.us[8 + i] = prow[8 + kb2 + i];

            // ---- O += P * V  (4 dim-tiles of 16) ----
#pragma unroll
            for (int t = 0; t < 4; ++t) {
                Frag16 bv;
#pragma unroll
                for (int i = 0; i < 16; ++i)
                    bv.us[i] = sV[(kb2 + i) * 64 + t * 16 + lm];
                if      (t == 0) o0 = wmma_bf16(pf, bv, o0);
                else if (t == 1) o1 = wmma_bf16(pf, bv, o1);
                else if (t == 2) o2 = wmma_bf16(pf, bv, o2);
                else             o3 = wmma_bf16(pf, bv, o3);
            }
        }
    }

    // ---- epilogue: normalize, stage in LDS, coalesced b128 stores ----
    unsigned short* sOw     = sO[wave];
    unsigned short* so_lane = sOw + hi * 8 * 64 + lm;
#pragma unroll
    for (int r = 0; r < 8; ++r) {
        float inv = 1.0f / lrun[r];
        so_lane[r * 64 + 0]  = f2bf(o0[r] * inv);
        so_lane[r * 64 + 16] = f2bf(o1[r] * inv);
        so_lane[r * 64 + 32] = f2bf(o2[r] * inv);
        so_lane[r * 64 + 48] = f2bf(o3[r] * inv);
    }
    __builtin_amdgcn_wave_barrier();
    asm volatile("" ::: "memory");
#pragma unroll
    for (int j = 0; j < 4; ++j) {
        int cid = lane + 32 * j;          // 0..127
        int tl  = cid >> 3;               // token-local 0..15
        int ch  = cid & 7;                // 16B chunk in 128B row
        uint4 d = *(const uint4*)(sOw + tl * 64 + ch * 8);
        *(uint4*)(Ob + ((size_t)b * SEQ + (q0 + tl)) * VDIM + h * HS + ch * 8) = d;
    }
}

// ---------------------------------------------------------------------------
// Kernel 3: out = O @ W_out^T.  M=8192, N=1024, K=1024, bf16 WMMA, fp32 out.
// One wave = 16(M) x 64(N); A fragment reused across 4 N-fragments.
// ---------------------------------------------------------------------------
__global__ __launch_bounds__(256) void out_proj_kernel(
        const unsigned short* __restrict__ Ob, const unsigned short* __restrict__ Wb,
        float* __restrict__ out) {
    __shared__ __align__(16) float sO3[8][16 * 64];   // 32 KB store staging

    int tid  = threadIdx.x;
    int lane = tid & 31;
    int wave = tid >> 5;
    int task = blockIdx.x * 8 + wave;    // 512 Mtiles * 16 Ngroups = 8192
    int ng   = task & 15;
    int mt   = task >> 4;
    int m0   = mt * 16;
    int n0   = ng * 64;

    int lm  = lane & 15;
    int hi  = lane >> 4;
    int kb  = hi ? 8 : 0;
    int kb2 = hi ? 16 : 0;

    v8f acc[4];
#pragma unroll
    for (int t = 0; t < 4; ++t)
#pragma unroll
        for (int i = 0; i < 8; ++i) acc[t][i] = 0.0f;

    const unsigned short* arow = Ob + (size_t)(m0 + lm) * VDIM;

    for (int k = 0; k < VDIM; k += 32) {
        if (k + 128 < VDIM)
            __builtin_prefetch(arow + k + 128, 0, 1);   // global_prefetch_b8

        Frag16 a;
#pragma unroll
        for (int i = 0; i < 8; ++i) a.us[i]     = arow[k + kb2 + i];
#pragma unroll
        for (int i = 0; i < 8; ++i) a.us[8 + i] = arow[k + kb2 + 8 + i];

#pragma unroll
        for (int t = 0; t < 4; ++t) {
            Frag16 bfg;
            const unsigned short* wrow = Wb + (size_t)(n0 + t * 16 + lm) * VDIM + k;
#pragma unroll
            for (int i = 0; i < 16; ++i) bfg.us[i] = wrow[kb2 + i];
            acc[t] = wmma_bf16(a, bfg, acc[t]);
        }
    }

    // stage fp32 tile in LDS, then coalesced float4 stores
    float* sw      = sO3[wave];
    float* sw_lane = sw + hi * 8 * 64 + lm;
#pragma unroll
    for (int t = 0; t < 4; ++t)
#pragma unroll
        for (int r = 0; r < 8; ++r)
            sw_lane[r * 64 + t * 16] = acc[t][r];

    __builtin_amdgcn_wave_barrier();
    asm volatile("" ::: "memory");
#pragma unroll
    for (int j = 0; j < 8; ++j) {
        int cid = lane + 32 * j;          // 0..255
        int row = cid >> 4;               // 0..15
        int ch  = cid & 15;               // 16B chunk in 256B row
        float4 d = *(const float4*)(sw + row * 64 + ch * 4);
        *(float4*)(out + (size_t)(m0 + row) * VDIM + n0 + ch * 4) = d;
    }
}

// ---------------------------------------------------------------------------
extern "C" void kernel_launch(void* const* d_in, const int* in_sizes, int n_in,
                              void* d_out, int out_size, void* d_ws, size_t ws_size,
                              hipStream_t stream) {
    const float* x    = (const float*)d_in[0];   // [4,2048,1024] fp32
    const float* Wqkv = (const float*)d_in[1];   // [192,64] fp32
    const float* Wout = (const float*)d_in[2];   // [1024,1024] fp32
    float*       out  = (float*)d_out;           // [4,2048,1024] fp32

    const size_t SZ = (size_t)TOKENS * VDIM;     // 8.39M elems per bf16 tensor
    unsigned short* Qb = (unsigned short*)d_ws;
    unsigned short* Kb = Qb + SZ;
    unsigned short* Vb = Kb + SZ;
    unsigned short* Ob = Vb + SZ;
    unsigned short* Wb = Ob + SZ;                // VDIM*VDIM bf16

    (void)in_sizes; (void)n_in; (void)out_size; (void)ws_size;

    cvt_bf16_kernel<<<1024, 256, 0, stream>>>(Wout, Wb, VDIM * VDIM);
    qkv_proj_kernel<<<1024, 256, 0, stream>>>(x, Wqkv, Qb, Kb, Vb);
    attn_kernel<<<BATCH * NHEADS * (SEQ / 64), 128, 0, stream>>>(Qb, Kb, Vb, Ob);
    out_proj_kernel<<<1024, 256, 0, stream>>>(Ob, Wb, out);
    (void)out;
}
